// SensorEncoder_23545010717434
// MI455X (gfx1250) — matrix operand order repaired and checked by reading, not compiled, not run on hardware
//
#include <hip/hip_runtime.h>
#include <math.h>

// ---------------------------------------------------------------------------
// CDNA5 WMMA types
// ---------------------------------------------------------------------------
typedef __attribute__((ext_vector_type(16))) _Float16 v16h;
typedef __attribute__((ext_vector_type(8)))  _Float16 v8h;
typedef __attribute__((ext_vector_type(8)))  float    v8f;

#define WMMA_F32_F16(a, b, c) \
  __builtin_amdgcn_wmma_f32_16x16x32_f16(false, (a), false, (b), (short)0, (c), false, false)

static __device__ __forceinline__ v16h cat8(v8h lo, v8h hi) {
  return __builtin_shufflevector(lo, hi, 0,1,2,3,4,5,6,7,8,9,10,11,12,13,14,15);
}

// Model constants
#define BB   32
#define SS   512
#define TT   513          // S + 1 (cls token)
#define DD   768
#define HH   8
#define HDX  96
#define LL   6
#define DFFX 3072
#define MM   16416        // BB * TT
#define TPAD 528          // padded token stride for K^T (multiple of 16, >= 527+1)

// ---------------------------------------------------------------------------
// Embedding
// ---------------------------------------------------------------------------
__global__ __launch_bounds__(256) void se_embed_kernel(
    const int* __restrict__ sensor_ids, const int* __restrict__ state_ids,
    const int* __restrict__ room_ids,   const float* __restrict__ coordinates,
    const float* __restrict__ time_deltas,
    const float* __restrict__ emb_sensor, const float* __restrict__ emb_state,
    const float* __restrict__ emb_room,   const float* __restrict__ fourier_W,
    const float* __restrict__ fourier_b,  const float* __restrict__ time_emb,
    const float* __restrict__ cls_token,  const float* __restrict__ pos_emb,
    float* __restrict__ x)
{
  int t = blockIdx.x;          // 0..512
  int b = blockIdx.y;
  int tid = threadIdx.x;
  __shared__ float ff[48];

  if (t > 0 && tid < 48) {
    int s = t - 1;
    int c = tid / 24;
    int rem = tid % 24;
    int iscos = rem / 12;
    int j = rem % 12;
    float coord = coordinates[((size_t)b * SS + s) * 2 + c];
    float ang = coord * exp2f((float)j) * 3.14159265358979f;
    ff[tid] = iscos ? cosf(ang) : sinf(ang);
  }
  __syncthreads();

  float* xr = x + ((size_t)b * TT + t) * DD;
  if (t == 0) {
    for (int d = tid; d < DD; d += 256)
      xr[d] = cls_token[d] + pos_emb[d];
    return;
  }
  int s = t - 1;
  int sid  = sensor_ids[b * SS + s];
  int stid = state_ids[b * SS + s];
  int rid  = room_ids[b * SS + s];
  float td = fmaxf(time_deltas[b * SS + s], 1e-6f);
  int idx = (int)(logf(td + 1.0f) * (1.0f / logf(3601.0f)) * 99.0f);
  idx = min(max(idx, 0), 99);

  for (int d = tid; d < DD; d += 256) {
    float v = emb_sensor[(size_t)sid * DD + d]
            + emb_state[(size_t)stid * DD + d]
            + emb_room[(size_t)rid * DD + d];
    float fsum = fourier_b[d];
    #pragma unroll 8
    for (int f = 0; f < 48; ++f) fsum += ff[f] * fourier_W[(size_t)f * DD + d];
    v += fsum + time_emb[(size_t)idx * DD + d] + pos_emb[(size_t)t * DD + d];
    xr[d] = v;
  }
}

// ---------------------------------------------------------------------------
// LayerNorm over D=768, one row per 256-thread block
// ---------------------------------------------------------------------------
__global__ __launch_bounds__(256) void se_ln_kernel(
    const float* __restrict__ in, const float* __restrict__ g,
    const float* __restrict__ bb, float* __restrict__ out)
{
  int row = blockIdx.x;
  int tid = threadIdx.x;
  const float* xr = in + (size_t)row * DD;
  float v0 = xr[tid], v1 = xr[tid + 256], v2 = xr[tid + 512];
  __shared__ float red[256];
  red[tid] = v0 + v1 + v2;
  __syncthreads();
  #pragma unroll
  for (int st = 128; st > 0; st >>= 1) {
    if (tid < st) red[tid] += red[tid + st];
    __syncthreads();
  }
  float mean = red[0] * (1.0f / 768.0f);
  __syncthreads();
  float d0 = v0 - mean, d1 = v1 - mean, d2 = v2 - mean;
  red[tid] = d0 * d0 + d1 * d1 + d2 * d2;
  __syncthreads();
  #pragma unroll
  for (int st = 128; st > 0; st >>= 1) {
    if (tid < st) red[tid] += red[tid + st];
    __syncthreads();
  }
  float rstd = rsqrtf(red[0] * (1.0f / 768.0f) + 1e-5f);
  float* orow = out + (size_t)row * DD;
  orow[tid]       = d0 * rstd * g[tid]       + bb[tid];
  orow[tid + 256] = d1 * rstd * g[tid + 256] + bb[tid + 256];
  orow[tid + 512] = d2 * rstd * g[tid + 512] + bb[tid + 512];
}

// ---------------------------------------------------------------------------
// Tiled WMMA GEMM, BK=64, double-buffered LDS, software-pipelined global loads.
// C = A[M,K] @ W[K,N] + bias (+residual, +gelu)
// outMode: 0 = f32 [M,N]; 1 = f16 scatter [B,H,T,96]; 2 = f16 scatter [B,H,96,TPAD]
// 16 WMMAs per barrier round; K assumed multiple of 64 (768 / 3072).
// ---------------------------------------------------------------------------
__global__ __launch_bounds__(256) void se_gemm_kernel(
    const float* __restrict__ A, const float* __restrict__ W,
    const float* __restrict__ bias, void* __restrict__ Cout,
    const float* __restrict__ residual,
    int M, int N, int K, int doGelu, int outMode)
{
  __shared__ _Float16 As[2][128 * 64];   // 2 x 16KB
  __shared__ _Float16 Bs[2][64 * 128];   // 2 x 16KB
  int tid  = threadIdx.x;
  int lane = tid & 31;
  int wave = tid >> 5;
  int wm = wave >> 2;     // 0..1  (64 rows)
  int wn = wave & 3;      // 0..3  (32 cols)
  int m0 = blockIdx.y * 128;
  int n0 = blockIdx.x * 128;

  // staging geometry: A 128x64 (2 thr/row, 32 cols each); B 64x128 (4 thr/row)
  int arow = tid >> 1;                 // 0..127
  int acs  = (tid & 1) * 32;           // 0/32
  int brow = tid >> 2;                 // 0..63
  int bcs  = (tid & 3) * 32;           // 0..96
  int gm_ld = m0 + arow;
  bool aval = gm_ld < M;
  const float4* abase = (const float4*)(A + (size_t)(aval ? gm_ld : 0) * K + acs);
  const float4* bbase = (const float4*)(W + (size_t)brow * N + n0 + bcs);
  const size_t bstep4 = (size_t)(N >> 2);   // float4s per K row

  v8f acc[4][2];
  #pragma unroll
  for (int mi = 0; mi < 4; ++mi)
    #pragma unroll
    for (int nj = 0; nj < 2; ++nj)
      #pragma unroll
      for (int e = 0; e < 8; ++e) acc[mi][nj][e] = 0.0f;

  float4 ra[8], rb[8];

  // ---- tile loader (global -> regs): 32 floats A + 32 floats B / thread ----
  auto load_tile = [&](int k0) {
    if (aval) {
      const float4* p = abase + (k0 >> 2);
      #pragma unroll
      for (int i = 0; i < 8; ++i) ra[i] = p[i];
    } else {
      #pragma unroll
      for (int i = 0; i < 8; ++i) { ra[i].x = 0.f; ra[i].y = 0.f; ra[i].z = 0.f; ra[i].w = 0.f; }
    }
    const float4* p = bbase + (size_t)k0 * bstep4;
    #pragma unroll
    for (int i = 0; i < 8; ++i) rb[i] = p[i];
  };
  // ---- regs -> LDS (f32 -> f16 pack) ----
  auto store_tile = [&](int buf) {
    #pragma unroll
    for (int i = 0; i < 4; ++i) {
      v8h hA, hB;
      hA[0]=(_Float16)ra[2*i].x;   hA[1]=(_Float16)ra[2*i].y;
      hA[2]=(_Float16)ra[2*i].z;   hA[3]=(_Float16)ra[2*i].w;
      hA[4]=(_Float16)ra[2*i+1].x; hA[5]=(_Float16)ra[2*i+1].y;
      hA[6]=(_Float16)ra[2*i+1].z; hA[7]=(_Float16)ra[2*i+1].w;
      *(v8h*)&As[buf][arow * 64 + acs + 8 * i] = hA;
      hB[0]=(_Float16)rb[2*i].x;   hB[1]=(_Float16)rb[2*i].y;
      hB[2]=(_Float16)rb[2*i].z;   hB[3]=(_Float16)rb[2*i].w;
      hB[4]=(_Float16)rb[2*i+1].x; hB[5]=(_Float16)rb[2*i+1].y;
      hB[6]=(_Float16)rb[2*i+1].z; hB[7]=(_Float16)rb[2*i+1].w;
      *(v8h*)&Bs[buf][brow * 128 + bcs + 8 * i] = hB;
    }
  };
  // ---- WMMA compute on one LDS stage (2 K-chunks, 16 WMMAs) ----
  auto compute = [&](int buf) {
    int hoff = (lane < 16) ? 0 : 8;
    #pragma unroll
    for (int kk = 0; kk < 2; ++kk) {
      v16h af[4], bf[2];
      #pragma unroll
      for (int mi = 0; mi < 4; ++mi) {
        int base = (wm * 64 + mi * 16 + (lane & 15)) * 64 + kk * 32 + hoff;
        v8h lo = *(const v8h*)&As[buf][base];
        v8h hi = *(const v8h*)&As[buf][base + 16];
        af[mi] = cat8(lo, hi);
      }
      #pragma unroll
      for (int nj = 0; nj < 2; ++nj) {
        #pragma unroll
        for (int s = 0; s < 16; ++s) {
          int kr = (s & 7) + ((s >> 3) << 4) + ((lane >> 4) << 3);
          bf[nj][s] = Bs[buf][(kk * 32 + kr) * 128 + wn * 32 + nj * 16 + (lane & 15)];
        }
      }
      #pragma unroll
      for (int mi = 0; mi < 4; ++mi)
        #pragma unroll
        for (int nj = 0; nj < 2; ++nj)
          acc[mi][nj] = WMMA_F32_F16(af[mi], bf[nj], acc[mi][nj]);
    }
  };

  // ---- software pipeline over K (BK = 64) ----
  load_tile(0);
  store_tile(0);
  __syncthreads();
  int buf = 0;
  for (int k0 = 0; k0 < K; k0 += 64) {
    bool more = (k0 + 64) < K;
    if (more) load_tile(k0 + 64);     // global loads overlap the 16 WMMAs below
    compute(buf);
    __syncthreads();
    if (more) {
      store_tile(buf ^ 1);
      __syncthreads();
    }
    buf ^= 1;
  }

  // ---- epilogue ----
  #pragma unroll
  for (int mi = 0; mi < 4; ++mi) {
    #pragma unroll
    for (int nj = 0; nj < 2; ++nj) {
      int gn = n0 + wn * 32 + nj * 16 + (lane & 15);
      #pragma unroll
      for (int e = 0; e < 8; ++e) {
        int gm = m0 + wm * 64 + mi * 16 + ((lane < 16) ? e : e + 8);
        if (gm >= M) continue;
        float val = acc[mi][nj][e] + bias[gn];
        if (residual) val += residual[(size_t)gm * N + gn];
        if (doGelu)   val = 0.5f * val * (1.0f + erff(val * 0.70710678118f));
        if (outMode == 0) {
          ((float*)Cout)[(size_t)gm * N + gn] = val;
        } else {
          int b = gm / TT, t = gm % TT;
          int h = gn / HDX, hd = gn % HDX;
          if (outMode == 1)
            ((_Float16*)Cout)[(((size_t)b * HH + h) * TT + t) * HDX + hd] = (_Float16)val;
          else
            ((_Float16*)Cout)[(((size_t)b * HH + h) * HDX + hd) * TPAD + t] = (_Float16)val;
        }
      }
    }
  }
}

// ---------------------------------------------------------------------------
// Flash-attention: one wave32 per (b, h, 16-row q tile).
// Q [B,H,T,96] f16, K^T [B,H,96,TPAD] f16, V [B,H,T,96] f16 -> out [B,T,768] f32.
// Q/P fragments: two aligned 16B vector loads; K/V: 16 scalar loads. Mask via one byte/lane+ballot.
// ---------------------------------------------------------------------------
__global__ __launch_bounds__(32) void se_attn_kernel(
    const _Float16* __restrict__ qg, const _Float16* __restrict__ kTg,
    const _Float16* __restrict__ vg, const unsigned char* __restrict__ mask,
    float* __restrict__ out)
{
  int qt0  = blockIdx.x * 16;
  int h    = blockIdx.y;
  int b    = blockIdx.z;
  int lane = threadIdx.x;
  const _Float16* qh = qg  + (size_t)(b * HH + h) * TT * HDX;
  const _Float16* kT = kTg + (size_t)(b * HH + h) * HDX * TPAD;
  const _Float16* vh = vg  + (size_t)(b * HH + h) * TT * HDX;
  float slope = exp2f(-(float)(h + 1));   // ALiBi slopes, H=8

  __shared__ float     sc[16][32];
  __shared__ _Float16  pf[16][32];
  __shared__ float     srow[16];

  int hoff = (lane < 16) ? 0 : 1;         // v8h index offset for A-layout halves

  // Q as three A-fragments (16x96 = 3 chunks of K=32); clamp row (stores guarded)
  v16h aq[3];
  {
    int qrow = qt0 + (lane & 15);
    if (qrow > TT - 1) qrow = TT - 1;
    const v8h* q8 = (const v8h*)(qh + (size_t)qrow * HDX);
    #pragma unroll
    for (int c = 0; c < 3; ++c)
      aq[c] = cat8(q8[c * 4 + hoff], q8[c * 4 + 2 + hoff]);
  }

  v8f oacc[6];
  #pragma unroll
  for (int c = 0; c < 6; ++c)
    #pragma unroll
    for (int e = 0; e < 8; ++e) oacc[c][e] = 0.0f;

  int r  = lane >> 1;
  int cb = (lane & 1) * 16;
  float m_r = -3.0e38f, l_r = 0.0f;

  for (int k0 = 0; k0 < TT; k0 += 32) {
    // per-lane key validity -> 32-bit ballot word
    int tokl = k0 + lane;
    bool okl = (tokl < TT) && (tokl == 0 || mask[b * SS + tokl - 1] != 0);
    unsigned mball = (unsigned)__ballot(okl);

    // ---- S = Q @ K^T (two 16x16 col tiles, contraction 96) ----
    v8f s0, s1;
    #pragma unroll
    for (int e = 0; e < 8; ++e) { s0[e] = 0.0f; s1[e] = 0.0f; }
    #pragma unroll
    for (int c = 0; c < 3; ++c) {
      v16h bk0, bk1;
      #pragma unroll
      for (int s = 0; s < 16; ++s) {
        int kr = (s & 7) + ((s >> 3) << 4) + ((lane >> 4) << 3);
        const _Float16* kq = kT + (size_t)(c * 32 + kr) * TPAD + k0 + (lane & 15);
        bk0[s] = kq[0];     // tokens k0..k0+15
        bk1[s] = kq[16];    // tokens k0+16..k0+31
      }
      s0 = WMMA_F32_F16(aq[c], bk0, s0);
      s1 = WMMA_F32_F16(aq[c], bk1, s1);
    }
    // ---- scale + ALiBi + mask -> LDS ----
    const float rscl = 0.1020620726f;   // 1/sqrt(96)
    #pragma unroll
    for (int e = 0; e < 8; ++e) {
      int row = (lane < 16) ? e : e + 8;
      int col = lane & 15;
      int qp  = qt0 + row;
      int kp0 = k0 + col, kp1 = k0 + 16 + col;
      float v0 = s0[e] * rscl - slope * fabsf((float)(qp - kp0));
      float v1 = s1[e] * rscl - slope * fabsf((float)(qp - kp1));
      sc[row][col]      = ((mball >> col) & 1u)        ? v0 : -3.0e38f;
      sc[row][16 + col] = ((mball >> (16 + col)) & 1u) ? v1 : -3.0e38f;
    }
    __syncthreads();
    // ---- online softmax (lane pair per row) ----
    float lmax = -3.0e38f;
    #pragma unroll
    for (int j = 0; j < 16; ++j) lmax = fmaxf(lmax, sc[r][cb + j]);
    lmax = fmaxf(lmax, __shfl_xor(lmax, 1));
    float mnew = fmaxf(m_r, lmax);
    float lsum = 0.0f;
    #pragma unroll
    for (int j = 0; j < 16; ++j) {
      float p = __expf(sc[r][cb + j] - mnew);
      pf[r][cb + j] = (_Float16)p;
      lsum += p;
    }
    lsum += __shfl_xor(lsum, 1);
    float sf = __expf(m_r - mnew);
    l_r = l_r * sf + lsum;
    m_r = mnew;
    if ((lane & 1) == 0) srow[r] = sf;
    __syncthreads();
    // ---- rescale O ----
    float rs8[8];
    #pragma unroll
    for (int e = 0; e < 8; ++e) rs8[e] = srow[(lane < 16) ? e : e + 8];
    #pragma unroll
    for (int c = 0; c < 6; ++c)
      #pragma unroll
      for (int e = 0; e < 8; ++e) oacc[c][e] *= rs8[e];
    // ---- P fragment from LDS (A layout) ----
    const v8h* pp = (const v8h*)&pf[lane & 15][0];
    v16h ap = cat8(pp[hoff], pp[2 + hoff]);
    // ---- O += P @ V (6 N-tiles over HD=96) ----
    #pragma unroll
    for (int c = 0; c < 6; ++c) {
      v16h bv;
      #pragma unroll
      for (int s = 0; s < 16; ++s) {
        int vtok = k0 + (s & 7) + ((s >> 3) << 4) + ((lane >> 4) << 3);
        bool vvld = vtok < TT;
        _Float16 vvv = vh[(size_t)(vvld ? vtok : 0) * HDX + c * 16 + (lane & 15)];
        bv[s] = vvld ? vvv : (_Float16)0.0f;    // avoid NaN propagation through WMMA
      }
      oacc[c] = WMMA_F32_F16(ap, bv, oacc[c]);
    }
    __syncthreads();   // sc/pf reused next iteration
  }

  // ---- finalize ----
  if ((lane & 1) == 0) srow[r] = l_r;
  __syncthreads();
  float dinv[8];
  #pragma unroll
  for (int e = 0; e < 8; ++e) {
    float l = srow[(lane < 16) ? e : e + 8];
    dinv[e] = (l > 0.0f) ? 1.0f / l : 0.0f;
  }
  #pragma unroll
  for (int c = 0; c < 6; ++c) {
    #pragma unroll
    for (int e = 0; e < 8; ++e) {
      int row = (lane < 16) ? e : e + 8;
      int tok = qt0 + row;
      if (tok < TT)
        out[((size_t)b * TT + tok) * DD + h * HDX + c * 16 + (lane & 15)] =
            oacc[c][e] * dinv[e];
    }
  }
}

// ---------------------------------------------------------------------------
// pooled = 0.5*cls + 0.5*masked_mean(tokens)
// ---------------------------------------------------------------------------
__global__ __launch_bounds__(256) void se_pool_kernel(
    const float* __restrict__ hf, const unsigned char* __restrict__ mask,
    float* __restrict__ pooled)
{
  int b = blockIdx.x, tid = threadIdx.x;
  __shared__ float mw[SS];
  __shared__ float red[256];
  for (int s = tid; s < SS; s += 256) mw[s] = mask[b * SS + s] ? 1.0f : 0.0f;
  __syncthreads();
  float c = 0.0f;
  for (int s = tid; s < SS; s += 256) c += mw[s];
  red[tid] = c;
  __syncthreads();
  #pragma unroll
  for (int st = 128; st > 0; st >>= 1) {
    if (tid < st) red[tid] += red[tid + st];
    __syncthreads();
  }
  float cnt = fmaxf(red[0], 1.0f);
  for (int d = tid; d < DD; d += 256) {
    float sum = 0.0f;
    for (int s = 0; s < SS; ++s)
      sum += hf[((size_t)b * TT + 1 + s) * DD + d] * mw[s];
    pooled[(size_t)b * DD + d] =
        0.5f * hf[((size_t)b * TT) * DD + d] + 0.5f * sum / cnt;
  }
}

// pooled2 = pooled @ pool_W + pool_b  (32x768 @ 768x768)
__global__ __launch_bounds__(256) void se_dense768_kernel(
    const float* __restrict__ in, const float* __restrict__ Wp,
    const float* __restrict__ bp, float* __restrict__ outp)
{
  int b = blockIdx.y, tid = threadIdx.x;
  int n = blockIdx.x * 256 + tid;
  __shared__ float sh[DD];
  for (int d = tid; d < DD; d += 256) sh[d] = in[(size_t)b * DD + d];
  __syncthreads();
  float acc = bp[n];
  for (int kk = 0; kk < DD; ++kk) acc += sh[kk] * Wp[(size_t)kk * DD + n];
  outp[(size_t)b * DD + n] = acc;
}

// clip = normalize(pooled2 @ clip_W)  (32x768 @ 768x512)
__global__ __launch_bounds__(256) void se_clip_kernel(
    const float* __restrict__ in, const float* __restrict__ Wc,
    float* __restrict__ outc)
{
  int b = blockIdx.x, tid = threadIdx.x;
  __shared__ float sh[DD];
  __shared__ float red[256];
  for (int d = tid; d < DD; d += 256) sh[d] = in[(size_t)b * DD + d];
  __syncthreads();
  float c0 = 0.0f, c1 = 0.0f;
  for (int kk = 0; kk < DD; ++kk) {
    float s = sh[kk];
    c0 += s * Wc[(size_t)kk * 512 + tid];
    c1 += s * Wc[(size_t)kk * 512 + tid + 256];
  }
  red[tid] = c0 * c0 + c1 * c1;
  __syncthreads();
  #pragma unroll
  for (int st = 128; st > 0; st >>= 1) {
    if (tid < st) red[tid] += red[tid + st];
    __syncthreads();
  }
  float inv = rsqrtf(red[0]);
  outc[(size_t)b * 512 + tid]       = c0 * inv;
  outc[(size_t)b * 512 + tid + 256] = c1 * inv;
}

// ---------------------------------------------------------------------------
// Host orchestration
// ---------------------------------------------------------------------------
extern "C" void kernel_launch(void* const* d_in, const int* in_sizes, int n_in,
                              void* d_out, int out_size, void* d_ws, size_t ws_size,
                              hipStream_t stream)
{
  (void)in_sizes; (void)n_in; (void)out_size; (void)ws_size;
  const size_t M  = (size_t)MM;
  const size_t MD = M * DD;

  float* ws  = (float*)d_ws;
  float* x   = ws;                 // [M, D] f32
  float* h   = ws + MD;            // [M, D] f32
  float* big = ws + 2 * MD;        // 4*MD f32, time-shared with FFN hidden
  // attention buffers inside 'big':
  float*     o    = big;                                   // [M, D] f32
  _Float16*  qh   = (_Float16*)(big + MD);                 // [B,H,T,96]  f16
  _Float16*  kTh  = qh  + (size_t)BB * HH * TT * HDX;      // [B,H,96,TPAD] f16
  _Float16*  vhf  = kTh + (size_t)BB * HH * HDX * TPAD;    // [B,H,T,96]  f16
  float* ffb = big;                // [M, 3072] f32, reuses o/q/k/v region
  float* pooled  = ws + 6 * MD;    // [B, D]
  float* pooled2 = pooled + BB * DD;

  const int*   sensor_ids = (const int*)d_in[0];
  const int*   state_ids  = (const int*)d_in[1];
  const int*   room_ids   = (const int*)d_in[2];
  const float* coords     = (const float*)d_in[3];
  const float* tdel       = (const float*)d_in[4];
  const unsigned char* mask = (const unsigned char*)d_in[5];
  const float* emb_sensor = (const float*)d_in[6];
  const float* emb_state  = (const float*)d_in[7];
  const float* emb_room   = (const float*)d_in[8];
  const float* fourier_W  = (const float*)d_in[9];
  const float* fourier_b  = (const float*)d_in[10];
  const float* time_emb   = (const float*)d_in[11];
  const float* cls_token  = (const float*)d_in[12];
  const float* pos_emb    = (const float*)d_in[13];
  const float* ln1_g = (const float*)d_in[14];
  const float* ln1_b = (const float*)d_in[15];
  const float* Wq = (const float*)d_in[16];
  const float* bq = (const float*)d_in[17];
  const float* Wk = (const float*)d_in[18];
  const float* bk = (const float*)d_in[19];
  const float* Wv = (const float*)d_in[20];
  const float* bv = (const float*)d_in[21];
  const float* Wo = (const float*)d_in[22];
  const float* bo = (const float*)d_in[23];
  const float* ln2_g = (const float*)d_in[24];
  const float* ln2_b = (const float*)d_in[25];
  const float* W1 = (const float*)d_in[26];
  const float* b1 = (const float*)d_in[27];
  const float* W2 = (const float*)d_in[28];
  const float* b2 = (const float*)d_in[29];
  const float* lnf_g = (const float*)d_in[30];
  const float* lnf_b = (const float*)d_in[31];
  const float* pool_W = (const float*)d_in[32];
  const float* pool_b = (const float*)d_in[33];
  const float* clip_W = (const float*)d_in[34];

  se_embed_kernel<<<dim3(TT, BB), 256, 0, stream>>>(
      sensor_ids, state_ids, room_ids, coords, tdel,
      emb_sensor, emb_state, emb_room, fourier_W, fourier_b, time_emb,
      cls_token, pos_emb, x);

  const dim3 gemmD(6, 129);    // N=768,  M=16416 -> 129 M-tiles
  const dim3 gemmFF(24, 129);  // N=3072

  for (int l = 0; l < LL; ++l) {
    se_ln_kernel<<<(int)M, 256, 0, stream>>>(x, ln1_g + (size_t)l * DD,
                                             ln1_b + (size_t)l * DD, h);
    se_gemm_kernel<<<gemmD, 256, 0, stream>>>(h, Wq + (size_t)l * DD * DD,
        bq + (size_t)l * DD, qh, nullptr, (int)M, DD, DD, 0, 1);
    se_gemm_kernel<<<gemmD, 256, 0, stream>>>(h, Wk + (size_t)l * DD * DD,
        bk + (size_t)l * DD, kTh, nullptr, (int)M, DD, DD, 0, 2);
    se_gemm_kernel<<<gemmD, 256, 0, stream>>>(h, Wv + (size_t)l * DD * DD,
        bv + (size_t)l * DD, vhf, nullptr, (int)M, DD, DD, 0, 1);

    se_attn_kernel<<<dim3(33, HH, BB), 32, 0, stream>>>(qh, kTh, vhf, mask, o);

    se_gemm_kernel<<<gemmD, 256, 0, stream>>>(o, Wo + (size_t)l * DD * DD,
        bo + (size_t)l * DD, x, x, (int)M, DD, DD, 0, 0);

    se_ln_kernel<<<(int)M, 256, 0, stream>>>(x, ln2_g + (size_t)l * DD,
                                             ln2_b + (size_t)l * DD, h);
    se_gemm_kernel<<<gemmFF, 256, 0, stream>>>(h, W1 + (size_t)l * DD * DFFX,
        b1 + (size_t)l * DFFX, ffb, nullptr, (int)M, DFFX, DD, 1, 0);
    se_gemm_kernel<<<gemmD, 256, 0, stream>>>(ffb, W2 + (size_t)l * DFFX * DD,
        b2 + (size_t)l * DD, x, x, (int)M, DD, DFFX, 0, 0);
  }

  se_ln_kernel<<<(int)M, 256, 0, stream>>>(x, lnf_g, lnf_b, h);
  se_pool_kernel<<<BB, 256, 0, stream>>>(h, mask, pooled);
  se_dense768_kernel<<<dim3(3, BB), 256, 0, stream>>>(pooled, pool_W, pool_b, pooled2);
  se_clip_kernel<<<BB, 256, 0, stream>>>(pooled2, clip_W, (float*)d_out);
}
